// VSSBlock_19756849562190
// MI455X (gfx1250) — matrix operand order repaired and checked by reading, not compiled, not run on hardware
//
#include <hip/hip_runtime.h>
#include <hip/hip_bf16.h>

#define DMODEL  192
#define DINNER  384
#define DSTATE  16
#define DTRANK  12
#define DCONV   4
#define BSZ     4
#define LSEQ    4096
#define BTOT    8                 // fwd + rev sequences
#define MTOK    (BTOT*LSEQ)       // 32768 tokens
#define NXZ     (2*DINNER)        // 768
#define NXPROJ  (DTRANK+2*DSTATE) // 44
#define NXP_PAD 48
#define LN_EPS  1e-5f

typedef __attribute__((ext_vector_type(16))) _Float16 v16h;
typedef __attribute__((ext_vector_type(8)))  _Float16 v8h;
typedef __attribute__((ext_vector_type(8)))  float    v8f;

// ---------------------------------------------------------------------------
// K0: transpose weights to f16, N-major ([N][K]) so WMMA B columns are
// contiguous in K. W_xproj padded to 48 columns with zeros.
// ---------------------------------------------------------------------------
__global__ void prep_weights(const float* __restrict__ Win,
                             const float* __restrict__ Wxp,
                             const float* __restrict__ Wout,
                             _Float16* __restrict__ WtIn,
                             _Float16* __restrict__ WtXp,
                             _Float16* __restrict__ WtOut) {
  int idx = blockIdx.x * 256 + threadIdx.x;
  const int n0 = NXZ * DMODEL;        // 147456
  const int n1 = NXP_PAD * DINNER;    // 18432
  const int n2 = DMODEL * DINNER;     // 73728
  if (idx < n0) {
    int n = idx / DMODEL, k = idx % DMODEL;
    WtIn[idx] = (_Float16)Win[k * NXZ + n];
  } else if (idx < n0 + n1) {
    int i = idx - n0;
    int n = i / DINNER, k = i % DINNER;
    float v = (n < NXPROJ) ? Wxp[k * NXPROJ + n] : 0.f;
    WtXp[i] = (_Float16)v;
  } else if (idx < n0 + n1 + n2) {
    int i = idx - n0 - n1;
    int n = i / DINNER, k = i % DINNER;
    WtOut[i] = (_Float16)Wout[k * DMODEL + n];
  }
}

// ---------------------------------------------------------------------------
// K1: per-token LayerNorm over 192 channels (x is [B,C,L] channel-strided),
// emit f16 tokens for both the forward sequence (b) and reversed copy (b+4).
// One wave per token, 6 channels per lane, shfl_xor wave32 reduction.
// ---------------------------------------------------------------------------
__global__ void ln_build(const float* __restrict__ x,
                         const float* __restrict__ gamma,
                         const float* __restrict__ beta,
                         _Float16* __restrict__ xin) {
  int wave = threadIdx.x >> 5, lane = threadIdx.x & 31;
  int tok = blockIdx.x * 8 + wave;          // 0..16383
  int b = tok >> 12, l = tok & (LSEQ - 1);
  const float* xp = x + (size_t)b * DMODEL * LSEQ + l;
  float v[6], s = 0.f, s2 = 0.f;
#pragma unroll
  for (int j = 0; j < 6; ++j) {
    int c = j * 32 + lane;
    float t = xp[(size_t)c * LSEQ];
    v[j] = t; s += t; s2 += t * t;
  }
#pragma unroll
  for (int off = 16; off; off >>= 1) {
    s  += __shfl_xor(s,  off, 32);
    s2 += __shfl_xor(s2, off, 32);
  }
  float mu   = s * (1.f / DMODEL);
  float var  = s2 * (1.f / DMODEL) - mu * mu;
  float rstd = rsqrtf(var + LN_EPS);
  _Float16* fwd = xin + ((size_t)b * LSEQ + l) * DMODEL;
  _Float16* rev = xin + ((size_t)(b + BSZ) * LSEQ + (LSEQ - 1 - l)) * DMODEL;
#pragma unroll
  for (int j = 0; j < 6; ++j) {
    int c = j * 32 + lane;
    float xn = (v[j] - mu) * rstd * gamma[c] + beta[c];
    fwd[c] = (_Float16)xn;
    rev[c] = (_Float16)xn;
  }
}

// ---------------------------------------------------------------------------
// WMMA GEMM, register-blocked: each wave computes a (16*MB) x (16*NB) macro
// tile with MB*NB v8f accumulators. Per K=32 step: MB A-fragments + NB
// B-fragments feed MB*NB v_wmma_f32_16x16x32_f16 ops (A reused NB times,
// B reused MB times -> ~21 FLOP per loaded byte at MB=2,NB=4).
// Fragment layouts used here (A and B share contiguous k order, not the ISA table order):
//   A lane: m = mT+(lane&15); halves = K[k0+(lane<16?0:16) .. +15] (contiguous)
//   B lane: n = nT+(lane&15); halves = K[k0+(lane<16?0:16) .. +15] (contiguous)
//   C lane: VGPR i -> row mT + (lane>>4)*8 + i, col nT + (lane&15)
// ---------------------------------------------------------------------------
template <int MB, int NB, bool OUT_HALF>
__global__ void gemm_wmma_f16(const _Float16* __restrict__ A,
                              const _Float16* __restrict__ Bt,
                              void* __restrict__ Cv,
                              int M, int Npad, int K, int ldc) {
  int wave = threadIdx.x >> 5, lane = threadIdx.x & 31;
  int l16 = lane & 15, hi = lane >> 4;
  int nGroups = Npad / (16 * NB);
  int tile = blockIdx.x * 8 + wave;
  int mT = (tile / nGroups) * (16 * MB);
  int nT = (tile % nGroups) * (16 * NB);
  if (mT >= M) return;

  const _Float16* a0[MB];
  const _Float16* b0[NB];
#pragma unroll
  for (int mi = 0; mi < MB; ++mi)
    a0[mi] = A + (size_t)(mT + mi * 16 + l16) * K;
#pragma unroll
  for (int ni = 0; ni < NB; ++ni)
    b0[ni] = Bt + (size_t)(nT + ni * 16 + l16) * K;

  v8f acc[MB][NB] = {};
  for (int k0 = 0; k0 < K; k0 += 32) {
    v16h af[MB], bf[NB];
#pragma unroll
    for (int mi = 0; mi < MB; ++mi) {
      v8h alo = *(const v8h*)(a0[mi] + k0 + hi * 16);
      v8h ahi = *(const v8h*)(a0[mi] + k0 + hi * 16 + 8);
#pragma unroll
      for (int i = 0; i < 8; ++i) { af[mi][i] = alo[i]; af[mi][8 + i] = ahi[i]; }
    }
#pragma unroll
    for (int ni = 0; ni < NB; ++ni)
      bf[ni] = *(const v16h*)(b0[ni] + k0 + hi * 16);
    __builtin_prefetch(a0[0] + k0 + 32, 0, 0);   // global_prefetch_b8
    __builtin_prefetch(b0[0] + k0 + 32, 0, 0);
#pragma unroll
    for (int mi = 0; mi < MB; ++mi)
#pragma unroll
      for (int ni = 0; ni < NB; ++ni)
        acc[mi][ni] = __builtin_amdgcn_wmma_f32_16x16x32_f16(
            /*neg_a=*/false, af[mi], /*neg_b=*/false, bf[ni],
            /*c_mod=*/(short)0, acc[mi][ni],
            /*reuse_a=*/false, /*reuse_b=*/false);
  }
#pragma unroll
  for (int mi = 0; mi < MB; ++mi) {
    int row0 = mT + mi * 16 + hi * 8;
#pragma unroll
    for (int ni = 0; ni < NB; ++ni) {
      int col = nT + ni * 16 + l16;
      if (OUT_HALF) {
        _Float16* C = (_Float16*)Cv;
#pragma unroll
        for (int i = 0; i < 8; ++i)
          C[(size_t)(row0 + i) * ldc + col] = (_Float16)acc[mi][ni][i];
      } else {
        float* C = (float*)Cv;
#pragma unroll
        for (int i = 0; i < 8; ++i)
          C[(size_t)(row0 + i) * ldc + col] = acc[mi][ni][i];
      }
    }
  }
}

// ---------------------------------------------------------------------------
// K3: depthwise causal conv (width 4) + bias + SiLU. xr lives in cols 0..383
// of the xz buffer (stride 768).
// ---------------------------------------------------------------------------
__global__ void conv_silu(const _Float16* __restrict__ xz,
                          const float* __restrict__ cw,
                          const float* __restrict__ cb,
                          _Float16* __restrict__ xc) {
  size_t idx = (size_t)blockIdx.x * 256 + threadIdx.x;
  if (idx >= (size_t)MTOK * DINNER) return;
  int e = (int)(idx % DINNER);
  size_t t = idx / DINNER;
  int l = (int)(t & (LSEQ - 1));
  size_t bl0 = t - l;                  // start of this sequence
  float acc = cb[e];
#pragma unroll
  for (int j = 0; j < DCONV; ++j) {
    int ll = l - (DCONV - 1) + j;
    if (ll >= 0)
      acc += (float)xz[(bl0 + ll) * NXZ + e] * cw[e * DCONV + j];
  }
  float sg = 1.f / (1.f + __expf(-acc));
  xc[idx] = (_Float16)(acc * sg);
}

// ---------------------------------------------------------------------------
// K5: selective scan. grid = 8 batches x 3 channel groups, block = 128.
// One channel per thread; 16-state h in VGPRs; dt = softplus(dbl[0:12]@Wdt+b).
// The shared 44-float x-proj row is staged in LDS once per token per block.
// ---------------------------------------------------------------------------
__global__ void ssm_scan(const float* __restrict__ dbl,
                         const _Float16* __restrict__ xc,
                         const _Float16* __restrict__ xz,
                         const float* __restrict__ Wdt,
                         const float* __restrict__ bdt,
                         const float* __restrict__ Alog,
                         const float* __restrict__ Dp,
                         _Float16* __restrict__ yg) {
  __shared__ float srow[NXP_PAD];
  int b = blockIdx.x / 3;
  int e = (blockIdx.x % 3) * 128 + threadIdx.x;
  float wdt[DTRANK], Av[DSTATE], h[DSTATE];
#pragma unroll
  for (int r = 0; r < DTRANK; ++r) wdt[r] = Wdt[r * DINNER + e];
#pragma unroll
  for (int n = 0; n < DSTATE; ++n) {
    Av[n] = -__expf(Alog[e * DSTATE + n]);
    h[n] = 0.f;
  }
  float bd = bdt[e], Dv = Dp[e];
  size_t base = (size_t)b * LSEQ;
  for (int l = 0; l < LSEQ; ++l) {
    __syncthreads();
    if (threadIdx.x < NXP_PAD)
      srow[threadIdx.x] = dbl[(base + l) * NXP_PAD + threadIdx.x];
    __syncthreads();
    float dr = bd;
#pragma unroll
    for (int r = 0; r < DTRANK; ++r) dr += srow[r] * wdt[r];
    float dt = dr > 20.f ? dr : log1pf(__expf(dr));
    float xt = (float)xc[(base + l) * DINNER + e];
    float dtx = dt * xt;
    float y = 0.f;
#pragma unroll
    for (int n = 0; n < DSTATE; ++n) {
      h[n] = __expf(dt * Av[n]) * h[n] + dtx * srow[DTRANK + n];
      y += h[n] * srow[DTRANK + DSTATE + n];
    }
    float zv = (float)xz[(base + l) * NXZ + DINNER + e];
    float zs = zv / (1.f + __expf(-zv));          // silu(z)
    yg[(base + l) * DINNER + e] = (_Float16)((y + xt * Dv) * zs);
  }
}

// ---------------------------------------------------------------------------
// K7: out = 0.5*(fwd + rev[::-1]) + residual, back to [B,C,H,W].
// ---------------------------------------------------------------------------
__global__ void combine_out(const float* __restrict__ out2,
                            const float* __restrict__ x,
                            float* __restrict__ out) {
  size_t idx = (size_t)blockIdx.x * 256 + threadIdx.x;
  if (idx >= (size_t)BSZ * DMODEL * LSEQ) return;
  int l = (int)(idx % LSEQ);
  size_t bc = idx / LSEQ;
  int c = (int)(bc % DMODEL);
  int b = (int)(bc / DMODEL);
  float f = out2[((size_t)b * LSEQ + l) * DMODEL + c];
  float r = out2[((size_t)(b + BSZ) * LSEQ + (LSEQ - 1 - l)) * DMODEL + c];
  out[idx] = 0.5f * (f + r) + x[idx];
}

// ---------------------------------------------------------------------------
extern "C" void kernel_launch(void* const* d_in, const int* in_sizes, int n_in,
                              void* d_out, int out_size, void* d_ws, size_t ws_size,
                              hipStream_t stream) {
  (void)in_sizes; (void)n_in; (void)out_size; (void)ws_size;
  const float* x     = (const float*)d_in[0];
  const float* gamma = (const float*)d_in[1];
  const float* beta  = (const float*)d_in[2];
  const float* Win   = (const float*)d_in[3];
  const float* cw    = (const float*)d_in[4];
  const float* cb    = (const float*)d_in[5];
  const float* Wxp   = (const float*)d_in[6];
  const float* Wdt   = (const float*)d_in[7];
  const float* bdt   = (const float*)d_in[8];
  const float* Alog  = (const float*)d_in[9];
  const float* Dp    = (const float*)d_in[10];
  const float* Wout  = (const float*)d_in[11];

  char* ws = (char*)d_ws;
  size_t off = 0;
  auto alloc = [&](size_t bytes) -> void* {
    void* p = ws + off;
    off = (off + bytes + 255) & ~(size_t)255;
    return p;
  };
  _Float16* xin   = (_Float16*)alloc((size_t)MTOK * DMODEL * 2);
  _Float16* WtIn  = (_Float16*)alloc((size_t)NXZ * DMODEL * 2);
  _Float16* WtXp  = (_Float16*)alloc((size_t)NXP_PAD * DINNER * 2);
  _Float16* WtOut = (_Float16*)alloc((size_t)DMODEL * DINNER * 2);
  _Float16* xzbuf = (_Float16*)alloc((size_t)MTOK * NXZ * 2);
  _Float16* xcbuf = (_Float16*)alloc((size_t)MTOK * DINNER * 2);
  float*    dblb  = (float*)   alloc((size_t)MTOK * NXP_PAD * 4);
  _Float16* ygb   = (_Float16*)alloc((size_t)MTOK * DINNER * 2);
  float*    out2  = (float*)   alloc((size_t)MTOK * DMODEL * 4);

  // K0: weight transposes to f16 N-major (B operands for WMMA)
  int prepN = NXZ * DMODEL + NXP_PAD * DINNER + DMODEL * DINNER;
  prep_weights<<<(prepN + 255) / 256, 256, 0, stream>>>(Win, Wxp, Wout,
                                                        WtIn, WtXp, WtOut);
  // K1: LayerNorm + fwd/rev token build
  ln_build<<<(BSZ * LSEQ) / 8, 256, 0, stream>>>(x, gamma, beta, xin);
  // K2: xz = xin @ W_in   (32768 x 768 x 192), f16 out; 32x64 wave macro-tile
  gemm_wmma_f16<2, 4, true>
      <<<(MTOK / 32) * (NXZ / 64) / 8, 256, 0, stream>>>(
          xin, WtIn, xzbuf, MTOK, NXZ, DMODEL, NXZ);
  // K3: depthwise conv + SiLU
  conv_silu<<<(int)(((size_t)MTOK * DINNER + 255) / 256), 256, 0, stream>>>(
      xzbuf, cw, cb, xcbuf);
  // K4: dbl = xc @ W_xproj (32768 x 48 x 384), f32 out; 32x48 wave macro-tile
  gemm_wmma_f16<2, 3, false>
      <<<(MTOK / 32) * (NXP_PAD / 48) / 8, 256, 0, stream>>>(
          xcbuf, WtXp, dblb, MTOK, NXP_PAD, DINNER, NXP_PAD);
  // K5: sequential selective scan + gating
  ssm_scan<<<BTOT * 3, 128, 0, stream>>>(dblb, xcbuf, xzbuf, Wdt, bdt, Alog,
                                         Dp, ygb);
  // K6: out2 = ygated @ W_out (32768 x 192 x 384), f32 out; 32x64 macro-tile
  gemm_wmma_f16<2, 4, false>
      <<<(MTOK / 32) * (DMODEL / 64) / 8, 256, 0, stream>>>(
          ygb, WtOut, out2, MTOK, DMODEL, DINNER, DMODEL);
  // K7: merge fwd/rev + residual, layout back to [B,C,H,W]
  combine_out<<<(int)(((size_t)BSZ * DMODEL * LSEQ + 255) / 256), 256, 0,
                stream>>>(out2, x, (float*)d_out);
}
